// MultiHeadAttention_22703197127367
// MI455X (gfx1250) — compile-verified, matrix operand order checked by reading, not run on hardware
//
#include <hip/hip_runtime.h>

// ---- problem constants (match reference) ----
#define Bc   4
#define Sc   2048
#define Dc   1024
#define Hc   16
#define DKc  64
#define Mtot (Bc * Sc)          // 8192

typedef __attribute__((ext_vector_type(16))) _Float16 v16h;
typedef __attribute__((ext_vector_type(8)))  float    v8f;

// K index of fragment element e for lane (16x16x32 f16 WMMA A/B layout, wave32)
__device__ __forceinline__ int amap_k(int e, int lane) {
    int hi = (lane & 16) ? 8 : 0;
    return (e < 8) ? (e + hi) : (e + 8 + hi);
}

// ---- gfx1250 async global->LDS copy (16B per lane) + ASYNCcnt waits ----
__device__ __forceinline__ void async_ld_b128(unsigned lds_off, const void* sbase,
                                              unsigned byte_off) {
    // GVS mode: 64-bit SGPR base + 32-bit per-lane VGPR byte offset.
    // VDST = per-lane LDS byte address.
    asm volatile("global_load_async_to_lds_b128 %0, %1, %2 offset:0"
                 :: "v"(lds_off), "v"(byte_off), "s"(sbase)
                 : "memory");
}

template <int N>
__device__ __forceinline__ void wait_asynccnt() {
#if __has_builtin(__builtin_amdgcn_s_wait_asynccnt)
    __builtin_amdgcn_s_wait_asynccnt(N);
#else
    asm volatile("s_wait_asynccnt %0" :: "i"(N) : "memory");
#endif
}

// =====================================================================
// f32 -> f16 elementwise convert (n divisible by 4)
// =====================================================================
__global__ void __launch_bounds__(256)
cvt_kernel(const float* __restrict__ in, _Float16* __restrict__ out, int n)
{
    int i = (blockIdx.x * 256 + threadIdx.x) * 4;
    if (i < n) {
        float4 v = *(const float4*)(in + i);
        out[i + 0] = (_Float16)v.x;
        out[i + 1] = (_Float16)v.y;
        out[i + 2] = (_Float16)v.z;
        out[i + 3] = (_Float16)v.w;
    }
}

// =====================================================================
// W[k,n] f32 -> Wt[n,k] f16 (LDS-tiled transpose, coalesced both sides)
// block (32,8); grid (Dc/32, Dc/32)
// =====================================================================
__global__ void __launch_bounds__(256)
cvt_transpose_kernel(const float* __restrict__ W, _Float16* __restrict__ Wt)
{
    __shared__ float t[32][33];
    int bx = blockIdx.x * 32;   // n-base
    int by = blockIdx.y * 32;   // k-base
    int tx = threadIdx.x, ty = threadIdx.y;
    #pragma unroll
    for (int i = 0; i < 32; i += 8)
        t[ty + i][tx] = W[(size_t)(by + ty + i) * Dc + bx + tx];   // t[k][n]
    __syncthreads();
    #pragma unroll
    for (int i = 0; i < 32; i += 8)
        Wt[(size_t)(bx + ty + i) * Dc + by + tx] = (_Float16)t[tx][ty + i];
}

// =====================================================================
// GEMM: Y = X @ W + b   (X f16 [M,Dc], Wt f16 [n,k] = W transposed)
//   OUT_HEADS: write f16 to [B,H,S,DK] head layout (else f32 [M,D])
// Block = 256 thr (8 waves), tile 64(M) x 128(N), staged K-step 64.
// Both tiles staged via double-buffered global_load_async_to_lds_b128
// (6 wave-issues per stage; s_wait_asynccnt 6 leaves only the next
// stage's copies in flight). Each wave: 32x32 patch, 8 wmma per stage.
// =====================================================================
template <bool OUT_HEADS>
__global__ void __launch_bounds__(256)
gemm_kernel(const _Float16* __restrict__ X,
            const _Float16* __restrict__ Wt,
            const float* __restrict__ bias,
            void* __restrict__ Out)
{
    __shared__ _Float16 lX[2][64][72];     // 144B row stride = 36 banks
    __shared__ _Float16 lWt[2][128][72];

    const int tid  = threadIdx.x;
    const int w    = tid >> 5;
    const int lane = tid & 31;
    const int wr   = w >> 2;           // 0..1
    const int wc   = w & 3;            // 0..3
    const int m0   = blockIdx.x * 64;
    const int n0   = blockIdx.y * 128;

    v8f acc[2][2] = {};

    // async staging: 16B chunks; X tile 64x64 halves = 512 chunks (2/thread),
    // Wt tile 128x64 halves = 1024 chunks (4/thread) -> 6 issues per wave/stage
    auto issue_stage = [&](int k0, int b) {
        #pragma unroll
        for (int i = 0; i < 2; ++i) {
            int idx = tid + 256 * i;
            int r = idx >> 3, c8 = (idx & 7) * 8;
            async_ld_b128((unsigned)(size_t)&lX[b][r][c8], X,
                          (unsigned)(((m0 + r) * Dc + k0 + c8) * 2));
        }
        #pragma unroll
        for (int i = 0; i < 4; ++i) {
            int idx = tid + 256 * i;
            int r = idx >> 3, c8 = (idx & 7) * 8;
            async_ld_b128((unsigned)(size_t)&lWt[b][r][c8], Wt,
                          (unsigned)(((n0 + r) * Dc + k0 + c8) * 2));
        }
    };

    const int nstages = Dc / 64;       // 16
    issue_stage(0, 0);

    for (int s = 0; s < nstages; ++s) {
        const int buf = s & 1;
        __syncthreads();   // all waves done reading buf^1 (stage s-1)
        if (s + 1 < nstages) {
            issue_stage((s + 1) * 64, buf ^ 1);
            wait_asynccnt<6>();        // only the 6 just-issued remain
        } else {
            wait_asynccnt<0>();
        }
        __syncthreads();   // stage s tiles visible block-wide

        #pragma unroll
        for (int kk = 0; kk < 2; ++kk) {
            v16h a[2], bf[2];
            #pragma unroll
            for (int mi = 0; mi < 2; ++mi) {
                int row = wr * 32 + mi * 16 + (lane & 15);
                #pragma unroll
                for (int e = 0; e < 16; ++e)
                    a[mi][e] = lX[buf][row][kk * 32 + amap_k(e, lane)];
            }
            #pragma unroll
            for (int ni = 0; ni < 2; ++ni) {
                int col = wc * 32 + ni * 16 + (lane & 15);
                #pragma unroll
                for (int e = 0; e < 16; ++e)
                    bf[ni][e] = lWt[buf][col][kk * 32 + amap_k(e, lane)];
            }
            #pragma unroll
            for (int mi = 0; mi < 2; ++mi)
                #pragma unroll
                for (int ni = 0; ni < 2; ++ni)
                    acc[mi][ni] = __builtin_amdgcn_wmma_f32_16x16x32_f16(
                        false, a[mi], false, bf[ni], (short)0, acc[mi][ni], false, false);
        }
    }

    // ---- epilogue ----
    const int hl = (lane & 16) ? 8 : 0;
    #pragma unroll
    for (int mi = 0; mi < 2; ++mi) {
        #pragma unroll
        for (int ni = 0; ni < 2; ++ni) {
            int colg = n0 + wc * 32 + ni * 16 + (lane & 15);
            float bv = bias[colg];
            #pragma unroll
            for (int e = 0; e < 8; ++e) {
                int rowg = m0 + wr * 32 + mi * 16 + e + hl;
                float v = acc[mi][ni][e] + bv;
                if (OUT_HEADS) {
                    int b = rowg / Sc, s = rowg % Sc;
                    int h = colg / DKc, dd = colg % DKc;
                    ((_Float16*)Out)[(((size_t)(b * Hc + h) * Sc) + s) * DKc + dd] = (_Float16)v;
                } else {
                    ((float*)Out)[(size_t)rowg * Dc + colg] = v;
                }
            }
        }
    }
}

// =====================================================================
// In-place RoPE on [B*H, S, 64] f16. One wave per row; lane l handles
// the rotate-half pair (l, l+32). Matches reference table exactly.
// =====================================================================
__global__ void __launch_bounds__(256)
rope_kernel(_Float16* __restrict__ X)
{
    int w = threadIdx.x >> 5, lane = threadIdx.x & 31;
    size_t row = (size_t)blockIdx.x * 8 + w;     // 0 .. B*H*S-1
    int s = (int)(row % Sc);
    _Float16* p = X + row * DKc;
    float x1 = (float)p[lane];
    float x2 = (float)p[lane + 32];
    // inv_freq = 10000^(-2*lane/64)  ->  exp(-(2*lane/64)*ln(10000))
    float inv = __expf(-((float)(2 * lane) / (float)DKc) * 9.210340371976184f);
    float fr = (float)s * inv;
    float c = __cosf(fr), sn = __sinf(fr);
    p[lane]      = (_Float16)(x1 * c - x2 * sn);
    p[lane + 32] = (_Float16)(x2 * c + x1 * sn);
}

// =====================================================================
// Flash-style causal attention. grid = (S/128, B*H), 256 thr (8 waves).
// K/V chunk tiles (32x64 f16) staged once per block via double-buffered
// global_load_async_to_lds_b128. S = Q K^T via 4 wmma; online softmax
// (shfl-xor over 16-lane halves); P transposed through wave-private
// LDS; O += P V via 4 wmma. Output f16 [B,S,D] merged-head layout.
// =====================================================================
__global__ void __launch_bounds__(256)
attn_kernel(const _Float16* __restrict__ q16,
            const _Float16* __restrict__ k16,
            const _Float16* __restrict__ v16,
            _Float16* __restrict__ attn)
{
    __shared__ _Float16 lK[2][32][72];
    __shared__ _Float16 lV[2][32][72];
    __shared__ _Float16 lP[8][16][36];

    const int tid = threadIdx.x;
    const int w = tid >> 5, lane = tid & 31;
    const int bh = blockIdx.y;
    const int bb = bh / Hc, hh = bh % Hc;
    const int qb = blockIdx.x * 128 + w * 16;
    const int hl = (lane & 16) ? 8 : 0;

    const _Float16* qp = q16 + (size_t)bh * Sc * DKc;
    const _Float16* kp = k16 + (size_t)bh * Sc * DKc;
    const _Float16* vp = v16 + (size_t)bh * Sc * DKc;

    const int skey = tid >> 3;          // 0..31
    const int sc8  = (tid & 7) * 8;     // 0..56

    v16h aq[2];
    {
        int r = qb + (lane & 15);
        #pragma unroll
        for (int ks = 0; ks < 2; ++ks)
            #pragma unroll
            for (int e = 0; e < 16; ++e)
                aq[ks][e] = qp[(size_t)r * DKc + ks * 32 + amap_k(e, lane)];
    }

    v8f o[4] = {};
    float mrow[8], lrow[8];
    #pragma unroll
    for (int e = 0; e < 8; ++e) { mrow[e] = -3.0e38f; lrow[e] = 0.f; }

    const int nchunks = blockIdx.x * 4 + 4;      // uniform per block

    {
        unsigned go = (unsigned)((0 + skey) * DKc + sc8) * 2u;
        async_ld_b128((unsigned)(size_t)&lK[0][skey][sc8], kp, go);
        async_ld_b128((unsigned)(size_t)&lV[0][skey][sc8], vp, go);
    }

    for (int kc = 0; kc < nchunks; ++kc) {
        const int kb  = kc * 32;
        const int buf = kc & 1;

        __syncthreads();
        if (kc + 1 < nchunks) {
            unsigned go = (unsigned)((kb + 32 + skey) * DKc + sc8) * 2u;
            async_ld_b128((unsigned)(size_t)&lK[buf ^ 1][skey][sc8], kp, go);
            async_ld_b128((unsigned)(size_t)&lV[buf ^ 1][skey][sc8], vp, go);
            wait_asynccnt<2>();
        } else {
            wait_asynccnt<0>();
        }
        __syncthreads();

        // ---- scores ----
        v8f sfr[2] = {};
        #pragma unroll
        for (int sub = 0; sub < 2; ++sub) {
            #pragma unroll
            for (int ks = 0; ks < 2; ++ks) {
                v16h bk;
                int key = sub * 16 + (lane & 15);
                #pragma unroll
                for (int e = 0; e < 16; ++e)
                    bk[e] = lK[buf][key][ks * 32 + amap_k(e, lane)];
                sfr[sub] = __builtin_amdgcn_wmma_f32_16x16x32_f16(
                    false, aq[ks], false, bk, (short)0, sfr[sub], false, false);
            }
        }

        // ---- scale + causal mask + online softmax ----
        float alpha[8];
        #pragma unroll
        for (int e = 0; e < 8; ++e) {
            int qi = qb + e + hl;
            #pragma unroll
            for (int sub = 0; sub < 2; ++sub) {
                int key = kb + sub * 16 + (lane & 15);
                float sc = sfr[sub][e] * 0.125f;
                if (key > qi) sc = -1.0e9f;
                sfr[sub][e] = sc;
            }
            float mx = fmaxf(sfr[0][e], sfr[1][e]);
            #pragma unroll
            for (int off = 1; off < 16; off <<= 1)
                mx = fmaxf(mx, __shfl_xor(mx, off, 32));
            float mnew = fmaxf(mrow[e], mx);
            alpha[e] = __expf(mrow[e] - mnew);
            mrow[e] = mnew;
            float ps = 0.f;
            #pragma unroll
            for (int sub = 0; sub < 2; ++sub) {
                float p = __expf(sfr[sub][e] - mnew);
                sfr[sub][e] = p;
                ps += p;
            }
            #pragma unroll
            for (int off = 1; off < 16; off <<= 1)
                ps += __shfl_xor(ps, off, 32);
            lrow[e] = lrow[e] * alpha[e] + ps;
        }
        #pragma unroll
        for (int nt = 0; nt < 4; ++nt)
            #pragma unroll
            for (int e = 0; e < 8; ++e)
                o[nt][e] *= alpha[e];

        // ---- transpose P via LDS ----
        __syncthreads();
        #pragma unroll
        for (int e = 0; e < 8; ++e) {
            int r = e + hl;
            lP[w][r][lane & 15]        = (_Float16)sfr[0][e];
            lP[w][r][(lane & 15) + 16] = (_Float16)sfr[1][e];
        }
        __syncthreads();
        v16h ap;
        #pragma unroll
        for (int e = 0; e < 16; ++e)
            ap[e] = lP[w][lane & 15][amap_k(e, lane)];

        // ---- O += P @ V ----
        #pragma unroll
        for (int nt = 0; nt < 4; ++nt) {
            v16h bv;
            #pragma unroll
            for (int e = 0; e < 16; ++e)
                bv[e] = lV[buf][amap_k(e, lane)][nt * 16 + (lane & 15)];
            o[nt] = __builtin_amdgcn_wmma_f32_16x16x32_f16(
                false, ap, false, bv, (short)0, o[nt], false, false);
        }
    }

    #pragma unroll
    for (int nt = 0; nt < 4; ++nt) {
        #pragma unroll
        for (int e = 0; e < 8; ++e) {
            int s = qb + e + hl;
            float v = o[nt][e] / lrow[e];
            attn[((size_t)bb * Sc + s) * Dc + hh * DKc + nt * 16 + (lane & 15)] = (_Float16)v;
        }
    }
}

// =====================================================================
extern "C" void kernel_launch(void* const* d_in, const int* in_sizes, int n_in,
                              void* d_out, int out_size, void* d_ws, size_t ws_size,
                              hipStream_t stream)
{
    const float* query = (const float*)d_in[0];
    const float* key   = (const float*)d_in[1];
    const float* value = (const float*)d_in[2];
    // d_in[3] = mask (causal tril) -- applied analytically in attn_kernel
    const float* Wq = (const float*)d_in[4];
    const float* bq = (const float*)d_in[5];
    const float* Wk = (const float*)d_in[6];
    const float* bk = (const float*)d_in[7];
    const float* Wv = (const float*)d_in[8];
    const float* bv = (const float*)d_in[9];
    const float* Wo = (const float*)d_in[10];
    const float* bo = (const float*)d_in[11];
    float* out = (float*)d_out;

    _Float16* ws = (_Float16*)d_ws;
    const size_t HS = (size_t)Bc * Hc * Sc * DKc;   // 8,388,608 halves
    const size_t MD = (size_t)Mtot * Dc;            // 8,388,608 halves
    const size_t WW = (size_t)Dc * Dc;              // 1,048,576 halves
    _Float16* q16    = ws;
    _Float16* k16    = q16 + HS;
    _Float16* v16    = k16 + HS;
    _Float16* attn16 = v16 + HS;                     // [M, D] f16
    _Float16* xbuf   = attn16 + MD;                  // reused f16 input buffer
    _Float16* wqT    = xbuf + MD;                    // [n,k] f16 transposed weights
    _Float16* wkT    = wqT + WW;
    _Float16* wvT    = wkT + WW;
    _Float16* woT    = wvT + WW;

    dim3 blk(256);
    dim3 tblk(32, 8), tgrid(Dc / 32, Dc / 32);
    cvt_transpose_kernel<<<tgrid, tblk, 0, stream>>>(Wq, wqT);
    cvt_transpose_kernel<<<tgrid, tblk, 0, stream>>>(Wk, wkT);
    cvt_transpose_kernel<<<tgrid, tblk, 0, stream>>>(Wv, wvT);
    cvt_transpose_kernel<<<tgrid, tblk, 0, stream>>>(Wo, woT);

    dim3 gCvt((unsigned)(MD / 1024));                // n/4 threads, 256/block
    dim3 gGemm(Mtot / 64, Dc / 128);                 // 128 x 8

    cvt_kernel<<<gCvt, blk, 0, stream>>>(query, xbuf, (int)MD);
    gemm_kernel<true><<<gGemm, blk, 0, stream>>>(xbuf, wqT, bq, q16);
    cvt_kernel<<<gCvt, blk, 0, stream>>>(key, xbuf, (int)MD);
    gemm_kernel<true><<<gGemm, blk, 0, stream>>>(xbuf, wkT, bk, k16);
    cvt_kernel<<<gCvt, blk, 0, stream>>>(value, xbuf, (int)MD);
    gemm_kernel<true><<<gGemm, blk, 0, stream>>>(xbuf, wvT, bv, v16);

    dim3 gRope(Bc * Hc * Sc / 8);                    // 16384
    rope_kernel<<<gRope, blk, 0, stream>>>(q16);
    rope_kernel<<<gRope, blk, 0, stream>>>(k16);

    dim3 gAttn(Sc / 128, Bc * Hc);                   // 16 x 64
    attn_kernel<<<gAttn, blk, 0, stream>>>(q16, k16, v16, attn16);

    gemm_kernel<false><<<gGemm, blk, 0, stream>>>(attn16, woT, bo, out);
}
